// OPTAttention_37434934952357
// MI455X (gfx1250) — matrix operand order repaired and checked by reading, not compiled, not run on hardware
//
#include <hip/hip_runtime.h>
#include <hip/hip_bf16.h>

// ---------------- constants (match reference) ----------------
#define TOTAL_T 4096      // B * SEQ
#define HID     4096
#define QKVN    12288
#define SEQ     2048
#define NHEAD   32
#define HDIM    128

typedef __attribute__((ext_vector_type(16))) _Float16 v16h;
typedef __attribute__((ext_vector_type(8)))  float    v8f;

union AFrag { v16h v; uint4 q[2]; };

static __device__ __forceinline__ v8f wmma_f16(v16h a, v16h b, v8f c) {
  // 8 args: (neg_a, A, neg_b, B, c_mod, C, reuse_a, reuse_b)
  return __builtin_amdgcn_wmma_f32_16x16x32_f16(false, a, false, b, (short)0, c,
                                                false, false);
}

static __device__ __forceinline__ v8f v8f_zero() {
  v8f z = {0.f,0.f,0.f,0.f,0.f,0.f,0.f,0.f};
  return z;
}

// ---- CDNA5 async global->LDS copy (16B per lane), tracked by ASYNCcnt ----
// Generic shared pointers keep the LDS byte offset in addr[31:0] (ISA 10.2).
static __device__ __forceinline__ void async_ld_b128(void* lds, const void* gptr) {
  unsigned loff = (unsigned)(unsigned long long)lds;
  asm volatile("global_load_async_to_lds_b128 %0, %1, off"
               :: "v"(loff), "v"(gptr) : "memory");
}
static __device__ __forceinline__ void async_wait0() {
  asm volatile("s_wait_asynccnt 0x0" ::: "memory");
}

// ---------------- f32 -> f16 conversion (flat) ----------------
__global__ __launch_bounds__(256)
void cvt_f16(const float* __restrict__ in, _Float16* __restrict__ out, long long n) {
  long long i = ((long long)blockIdx.x * blockDim.x + threadIdx.x) * 4;
  if (i + 3 < n) {
    float4 v = *(const float4*)(in + i);
    union { _Float16 h[4]; uint2 u; } r;
    r.h[0] = (_Float16)v.x; r.h[1] = (_Float16)v.y;
    r.h[2] = (_Float16)v.z; r.h[3] = (_Float16)v.w;
    *(uint2*)(out + i) = r.u;
  }
}

// ------------ f32 -> f16 conversion + transpose (for weights) ------------
// in[K][N] f32  ->  out[N][K] f16 ; 32x32 tiles via padded LDS, coalesced both ways.
__global__ __launch_bounds__(256)
void cvt_transpose_f16(const float* __restrict__ in, _Float16* __restrict__ out,
                       int K, int N) {
  __shared__ _Float16 t[32][33];
  const int tx = threadIdx.x, ty = threadIdx.y;
  const int n0 = blockIdx.x * 32, k0 = blockIdx.y * 32;
#pragma unroll
  for (int j = 0; j < 32; j += 8)
    t[ty + j][tx] = (_Float16)in[(size_t)(k0 + ty + j) * N + n0 + tx];
  __syncthreads();
#pragma unroll
  for (int j = 0; j < 32; j += 8)
    out[(size_t)(n0 + ty + j) * K + k0 + tx] = t[tx][ty + j];
}

// ---------------- tiled WMMA GEMM (double-buffered async staging) ----------------
// C[M x N] = A[M x K] * B[K x N] + bias, with B supplied TRANSPOSED as BT[N][K].
// Block tile 128x128, K-step 32; 8 waves = 2(M) x 4(N), each owns 64x32 (4x2 tiles).
// Both A and BT tiles staged with global_load_async_to_lds_b128, double-buffered.
// MODE 0: plain f32 store. MODE 1: QKV scatter epilogue (Q scaled by 1/sqrt(128)).
template<int MODE>
__global__ __launch_bounds__(256)
void gemm_wmma(const _Float16* __restrict__ A,    // [M][K]
               const _Float16* __restrict__ BT,   // [N][K]
               const float* __restrict__ bias, float* __restrict__ C,
               _Float16* __restrict__ Qo, _Float16* __restrict__ Ko,
               _Float16* __restrict__ Vo, int N, int Kdim) {
  __shared__ __align__(32) _Float16 As[2][128 * 32];
  __shared__ __align__(32) _Float16 Bs[2][128 * 32];

  const int tid  = threadIdx.x;
  const int lane = tid & 31;
  const int wave = tid >> 5;
  const int wm   = (wave >> 2) * 64;
  const int wn   = (wave & 3) * 32;
  const int m0   = blockIdx.x * 128;
  const int n0   = blockIdx.y * 128;
  const int lh   = lane >> 4;
  const int ll   = lane & 15;

  v8f acc[4][2];
#pragma unroll
  for (int mi = 0; mi < 4; ++mi)
#pragma unroll
    for (int ni = 0; ni < 2; ++ni) acc[mi][ni] = v8f_zero();

  const int arow = tid >> 1, acol = (tid & 1) * 16;   // 256 thr cover 128x32
  const _Float16* gA = A  + (size_t)(m0 + arow) * Kdim + acol;
  const _Float16* gB = BT + (size_t)(n0 + arow) * Kdim + acol;

  auto stage = [&](int k0, int buf) {
    _Float16* la = &As[buf][arow * 32 + acol];
    _Float16* lb = &Bs[buf][arow * 32 + acol];
    async_ld_b128(la,     gA + k0);
    async_ld_b128(la + 8, gA + k0 + 8);
    async_ld_b128(lb,     gB + k0);
    async_ld_b128(lb + 8, gB + k0 + 8);
  };

  stage(0, 0);
  async_wait0();
  __syncthreads();

  for (int k0 = 0; k0 < Kdim; k0 += 32) {
    const int cur = (k0 >> 5) & 1;
    if (k0 + 32 < Kdim) stage(k0 + 32, cur ^ 1);   // prefetch next tile (async)

    // A fragment (16-bit A 16x32): per lane contiguous K (lh*16 .. +15)
    AFrag af[4];
#pragma unroll
    for (int mi = 0; mi < 4; ++mi) {
      const _Float16* p = &As[cur][(wm + mi * 16 + ll) * 32 + lh * 16];
      af[mi].q[0] = *(const uint4*)p;
      af[mi].q[1] = *(const uint4*)(p + 8);
    }
    // B fragment: lane holds column n = ll, contiguous K (lh*16 .. +15)
    v16h bf[2];
#pragma unroll
    for (int ni = 0; ni < 2; ++ni)
      bf[ni] = *(const v16h*)&Bs[cur][(wn + ni * 16 + ll) * 32 + lh * 16];

#pragma unroll
    for (int mi = 0; mi < 4; ++mi)
#pragma unroll
      for (int ni = 0; ni < 2; ++ni)
        acc[mi][ni] = wmma_f16(af[mi].v, bf[ni], acc[mi][ni]);

    async_wait0();     // next tile's async copies have landed
    __syncthreads();   // everyone done reading cur; safe to overwrite next iter
  }

  float bias2[2];
  bias2[0] = bias[n0 + wn + ll];
  bias2[1] = bias[n0 + wn + 16 + ll];

#pragma unroll
  for (int mi = 0; mi < 4; ++mi)
#pragma unroll
    for (int ni = 0; ni < 2; ++ni)
#pragma unroll
      for (int i = 0; i < 8; ++i) {
        // C layout: VGPR i holds row (i + 8*half), col = lane&15 within tile
        int row = m0 + wm + mi * 16 + i + 8 * lh;
        int col = n0 + wn + ni * 16 + ll;
        float v = acc[mi][ni][i] + bias2[ni];
        if (MODE == 0) {
          C[(size_t)row * N + col] = v;
        } else {
          int which = col >> 12;          // 0=Q 1=K 2=V
          int c2 = col & 4095;
          int h = c2 >> 7, d = c2 & 127;
          int b = row >> 11, s = row & 2047;
          size_t idx = (((size_t)(b * NHEAD + h)) * SEQ + s) * HDIM + d;
          if (which == 0)      Qo[idx] = (_Float16)(v * 0.08838834764831845f);
          else if (which == 1) Ko[idx] = (_Float16)v;
          else                 Vo[idx] = (_Float16)v;
        }
      }
}

// ---------------- causal flash attention (WMMA, double-buffered) ----------------
// grid: (SEQ/64, NHEAD, B). 128 threads = 4 waves; wave w owns query rows
// q0+16w..+15. Key blocks of 32. K staged via async copy; V staged transposed.
__global__ __launch_bounds__(128)
void flash_attn_wmma(const _Float16* __restrict__ Q, const _Float16* __restrict__ K,
                     const _Float16* __restrict__ V, _Float16* __restrict__ O) {
  __shared__ __align__(32) _Float16 Ks[2][32 * 136];   // K rows, padded
  __shared__ __align__(32) _Float16 Vt[2][128 * 40];   // V^T: Vt[d][k], padded
  __shared__ __align__(32) _Float16 Ps[4][16 * 48];    // per-wave P tile, padded

  const int qt = blockIdx.x, h = blockIdx.y, b = blockIdx.z;
  const int tid = threadIdx.x, lane = tid & 31, wave = tid >> 5;
  const int lh = lane >> 4, ll = lane & 15;
  const size_t head = ((size_t)(b * NHEAD + h)) * SEQ * HDIM;
  const _Float16* Qh = Q + head;
  const _Float16* Kh = K + head;
  const _Float16* Vh = V + head;
  const int q0 = qt * 64;

  // Q fragments for this wave's 16 rows: 4 k-chunks of 32 head-dims
  const int qrow = q0 + wave * 16 + ll;
  AFrag qf[4];
#pragma unroll
  for (int dc = 0; dc < 4; ++dc) {
    const _Float16* p = Qh + (size_t)qrow * HDIM + dc * 32 + lh * 16;
    qf[dc].q[0] = *(const uint4*)p;
    qf[dc].q[1] = *(const uint4*)(p + 8);
  }

  v8f o[8];
#pragma unroll
  for (int n = 0; n < 8; ++n) o[n] = v8f_zero();
  float mrun[8], lrun[8];
#pragma unroll
  for (int i = 0; i < 8; ++i) { mrun[i] = -3.0e38f; lrun[i] = 0.f; }

  const int nkb = 2 * qt + 2;                 // causal: keys 0 .. q0+63
  const int srow = tid >> 2, schunk = (tid & 3) * 32;  // staging assignment

  auto stage_kv = [&](int kb, int buf) {
    { // K block 32x128 via async (4x 16B per thread)
      const _Float16* g = Kh + (size_t)(kb + srow) * HDIM + schunk;
      _Float16* l = &Ks[buf][srow * 136 + schunk];
      async_ld_b128(l,      g);
      async_ld_b128(l + 8,  g + 8);
      async_ld_b128(l + 16, g + 16);
      async_ld_b128(l + 24, g + 24);
    }
    { // V block transposed: coalesced global read, b16 scatter to Vt
      const _Float16* g = Vh + (size_t)(kb + srow) * HDIM + schunk;
      _Float16 tmp[32];
      *(uint4*)&tmp[0]  = *(const uint4*)g;
      *(uint4*)&tmp[8]  = *(const uint4*)(g + 8);
      *(uint4*)&tmp[16] = *(const uint4*)(g + 16);
      *(uint4*)&tmp[24] = *(const uint4*)(g + 24);
#pragma unroll
      for (int j = 0; j < 32; ++j) Vt[buf][(schunk + j) * 40 + srow] = tmp[j];
    }
  };

  stage_kv(0, 0);
  async_wait0();
  __syncthreads();

  for (int kb0 = 0; kb0 < nkb; ++kb0) {
    const int kb = kb0 * 32;
    const int cur = kb0 & 1;
    if (kb0 + 1 < nkb) stage_kv(kb + 32, cur ^ 1);   // prefetch next block

    // scores S = Q * K^T : two 16x16 tiles (key cols kb..+15, kb+16..+31)
    v8f s0 = v8f_zero(), s1 = v8f_zero();
#pragma unroll
    for (int dc = 0; dc < 4; ++dc) {
      AFrag k0f, k1f;
      const _Float16* b0 = &Ks[cur][ll * 136 + dc * 32 + lh * 16];
      const _Float16* b1 = &Ks[cur][(16 + ll) * 136 + dc * 32 + lh * 16];
      k0f.q[0] = *(const uint4*)b0; k0f.q[1] = *(const uint4*)(b0 + 8);
      k1f.q[0] = *(const uint4*)b1; k1f.q[1] = *(const uint4*)(b1 + 8);
      s0 = wmma_f16(qf[dc].v, k0f.v, s0);
      s1 = wmma_f16(qf[dc].v, k1f.v, s1);
    }

    const int rbase = q0 + wave * 16 + 8 * lh;  // row for acc index i is rbase+i
    const int c0 = kb + ll, c1 = c0 + 16;
    _Float16* pw = &Ps[wave][0];
#pragma unroll
    for (int i = 0; i < 8; ++i) {
      const int row = rbase + i;
      float a0 = (c0 <= row) ? s0[i] : -3.0e38f;   // causal mask
      float a1 = (c1 <= row) ? s1[i] : -3.0e38f;
      float mx = fmaxf(a0, a1);
      mx = fmaxf(mx, __shfl_xor(mx, 1, 32));
      mx = fmaxf(mx, __shfl_xor(mx, 2, 32));
      mx = fmaxf(mx, __shfl_xor(mx, 4, 32));
      mx = fmaxf(mx, __shfl_xor(mx, 8, 32));
      float mnew  = fmaxf(mrun[i], mx);
      float alpha = __expf(mrun[i] - mnew);
      float p0 = __expf(a0 - mnew);
      float p1 = __expf(a1 - mnew);
      float ps = p0 + p1;
      ps += __shfl_xor(ps, 1, 32);
      ps += __shfl_xor(ps, 2, 32);
      ps += __shfl_xor(ps, 4, 32);
      ps += __shfl_xor(ps, 8, 32);
      lrun[i] = lrun[i] * alpha + ps;
      mrun[i] = mnew;
#pragma unroll
      for (int n = 0; n < 8; ++n) o[n][i] *= alpha;   // rescale O rows
      int prow = i + 8 * lh;
      pw[prow * 48 + ll]      = (_Float16)p0;
      pw[prow * 48 + 16 + ll] = (_Float16)p1;
    }

    // reload P as a 16x32 A-fragment (wave-private LDS region; DS in-order)
    AFrag pf;
    pf.q[0] = *(const uint4*)&pw[ll * 48 + lh * 16];
    pf.q[1] = *(const uint4*)&pw[ll * 48 + lh * 16 + 8];

    // O += P * V : 8 head-dim tiles; V fragment = 2 contiguous b128 LDS reads
#pragma unroll
    for (int n = 0; n < 8; ++n) {
      AFrag vfu;
      const _Float16* vp = &Vt[cur][(n * 16 + ll) * 40 + lh * 16];
      vfu.q[0] = *(const uint4*)vp;
      vfu.q[1] = *(const uint4*)(vp + 8);
      o[n] = wmma_f16(pf.v, vfu.v, o[n]);
    }

    async_wait0();
    __syncthreads();
  }

  // normalize and scatter f16 attn output [T, HID]
#pragma unroll
  for (int n = 0; n < 8; ++n)
#pragma unroll
    for (int i = 0; i < 8; ++i) {
      int row = q0 + wave * 16 + i + 8 * lh;
      float val = o[n][i] / lrun[i];
      O[((size_t)b * SEQ + row) * HID + h * HDIM + n * 16 + ll] = (_Float16)val;
    }
}

// ---------------- launcher ----------------
extern "C" void kernel_launch(void* const* d_in, const int* in_sizes, int n_in,
                              void* d_out, int out_size, void* d_ws, size_t ws_size,
                              hipStream_t stream) {
  const float* hs    = (const float*)d_in[0];
  const float* w_qkv = (const float*)d_in[1];
  const float* b_qkv = (const float*)d_in[2];
  const float* w_o   = (const float*)d_in[3];
  const float* b_o   = (const float*)d_in[4];
  (void)in_sizes; (void)n_in; (void)out_size; (void)ws_size;

  char* ws = (char*)d_ws;
  _Float16* hs16    = (_Float16*)ws; ws += (size_t)TOTAL_T * HID * 2;
  _Float16* wqkv16T = (_Float16*)ws; ws += (size_t)HID * QKVN * 2;   // [QKVN][HID]
  _Float16* wo16T   = (_Float16*)ws; ws += (size_t)HID * HID * 2;    // [HID][HID]
  _Float16* Qb      = (_Float16*)ws; ws += (size_t)TOTAL_T * HID * 2;
  _Float16* Kb      = (_Float16*)ws; ws += (size_t)TOTAL_T * HID * 2;
  _Float16* Vb      = (_Float16*)ws; ws += (size_t)TOTAL_T * HID * 2;
  _Float16* attn16  = (_Float16*)ws;

  // 1) down-convert activations; down-convert + transpose weights
  {
    long long n1 = (long long)TOTAL_T * HID;
    cvt_f16<<<(unsigned)((n1 / 4 + 255) / 256), 256, 0, stream>>>(hs, hs16, n1);
    cvt_transpose_f16<<<dim3(QKVN / 32, HID / 32), dim3(32, 8), 0, stream>>>(
        w_qkv, wqkv16T, HID, QKVN);
    cvt_transpose_f16<<<dim3(HID / 32, HID / 32), dim3(32, 8), 0, stream>>>(
        w_o, wo16T, HID, HID);
  }

  // 2) QKV projection: [4096 x 4096] * [4096 x 12288] + bias -> Q,K,V f16
  gemm_wmma<1><<<dim3(TOTAL_T / 128, QKVN / 128), 256, 0, stream>>>(
      hs16, wqkv16T, b_qkv, nullptr, Qb, Kb, Vb, QKVN, HID);

  // 3) causal flash attention per (query-tile, head, batch)
  flash_attn_wmma<<<dim3(SEQ / 64, NHEAD, 2), 128, 0, stream>>>(Qb, Kb, Vb, attn16);

  // 4) output projection: [4096 x 4096] * [4096 x 4096] + bias -> f32 d_out
  gemm_wmma<0><<<dim3(TOTAL_T / 128, HID / 128), 256, 0, stream>>>(
      attn16, wo16T, b_o, (float*)d_out, nullptr, nullptr, nullptr, HID, HID);
}
